// SwinTransformerBlock3D_58617713655998
// MI455X (gfx1250) — matrix operand order repaired and checked by reading, not compiled, not run on hardware
//
#include <hip/hip_runtime.h>
#include <hip/hip_bf16.h>
#include <math.h>

typedef __attribute__((ext_vector_type(16))) _Float16 v16h;
typedef __attribute__((ext_vector_type(8)))  _Float16 v8h;
typedef __attribute__((ext_vector_type(8)))  float    v8f;

#define DIM 128
#define NH 4
#define HD 32
#define WS 7
#define SS 3
#define ZP 14
#define HP 56
#define WP 56
#define NTOK 343
#define NPAD 352            // 22 * 16
#define NT_TILES 22
#define NWIN 256            // B(2) * 2*8*8
#define NMLP 512
#define TOKENS 87808        // 2*14*56*56
#define MT_MLP 5488         // TOKENS/16
#define QK_SCALE 0.17677669529663687f

// ---------------- WMMA fragment helpers (layouts per CDNA5 ISA) ----

// A fragment: 16x32 f16, row-major source with leading dim lda (elements).
// lane l: row m = l&15, khalf = l>>4. VGPR0-3 hold k = khalf*8 + 0..7,
// VGPR4-7 hold k = 16 + khalf*8 + 0..7.
__device__ inline v16h load_a_frag(const _Float16* base, int lda, int lane) {
  int m = lane & 15;
  int kh = (lane >> 4) * 8;
  const _Float16* p = base + (size_t)m * lda + kh;
  v8h lo = *(const v8h*)(p);
  v8h hi = *(const v8h*)(p + 16);
  return __builtin_shufflevector(lo, hi, 0,1,2,3,4,5,6,7,8,9,10,11,12,13,14,15);
}

// B fragment: 32x16 f16 from a B^T (i.e. [N][K]) row-major source, ld = K stride.
// lane l: col n = l&15, kh = (l>>4)*8: k = kh + 0..7, then k = 16 + kh + 0..7.
__device__ inline v16h load_b_frag(const _Float16* baseT, int ldb, int lane) {
  int n = lane & 15;
  int kb = (lane >> 4) * 8;
  const _Float16* p = baseT + (size_t)n * ldb + kb;
  v8h lo = *(const v8h*)(p);
  v8h hi = *(const v8h*)(p + 16);
  return __builtin_shufflevector(lo, hi, 0,1,2,3,4,5,6,7,8,9,10,11,12,13,14,15);
}

__device__ inline v8f wmma16(v16h a, v16h b, v8f c) {
  return __builtin_amdgcn_wmma_f32_16x16x32_f16(false, a, false, b, (short)0, c,
                                                false, false);
}

__device__ inline v8f zero8() {
  v8f z = {0.f,0.f,0.f,0.f,0.f,0.f,0.f,0.f};
  return z;
}

// ---------------- weight transpose + f16 convert:  dst[f*K + c] = src[c*Nf + f]
__global__ void k_transpose_f16(const float* __restrict__ src,
                                _Float16* __restrict__ dst, int K, int Nf) {
  int idx = blockIdx.x * blockDim.x + threadIdx.x;
  if (idx >= K * Nf) return;
  int f = idx / K, c = idx % K;
  dst[idx] = (_Float16)src[c * Nf + f];
}

// ---------------- LN1 + cyclic shift + window partition -> xw f16 [256][352][128]
__global__ __launch_bounds__(256) void k_ln1(const float* __restrict__ x,
                                             const float* __restrict__ g,
                                             const float* __restrict__ b,
                                             _Float16* __restrict__ xw) {
  int wv = threadIdx.x >> 5, lane = threadIdx.x & 31;
  int token = blockIdx.x * 8 + wv;               // 0 .. 256*352-1
  int win = token / NPAD, n = token % NPAD;
  _Float16* dst = xw + (size_t)token * DIM;
  if (n >= NTOK) {
    #pragma unroll
    for (int i = 0; i < 4; ++i) dst[lane + 32 * i] = (_Float16)0.f;
    return;
  }
  int bb = win / 128, wrem = win % 128;
  int wz = wrem / 64, wh = (wrem / 8) % 8, ww = wrem % 8;
  int tz = n / 49, th = (n / 7) % 7, tw = n % 7;
  int z = (wz * WS + tz + SS) % ZP;
  int h = (wh * WS + th + SS) % HP;
  int w = (ww * WS + tw + SS) % WP;
  const float* xp = x + ((((size_t)bb * ZP + z) * HP + h) * WP + w) * DIM;
  float v0[4], s = 0.f, s2 = 0.f;
  #pragma unroll
  for (int i = 0; i < 4; ++i) { v0[i] = xp[lane + 32 * i]; s += v0[i]; s2 += v0[i] * v0[i]; }
  #pragma unroll
  for (int off = 16; off >= 1; off >>= 1) {
    s  += __shfl_xor(s,  off, 32);
    s2 += __shfl_xor(s2, off, 32);
  }
  float mean = s * (1.f / DIM);
  float var = s2 * (1.f / DIM) - mean * mean;
  float rstd = rsqrtf(var + 1e-5f);
  #pragma unroll
  for (int i = 0; i < 4; ++i) {
    int c = lane + 32 * i;
    dst[c] = (_Float16)((v0[i] - mean) * rstd * g[c] + b[c]);
  }
}

// ---------------- QKV GEMM: [352,128] x [128,384]; one mtile per wave ---------
__global__ __launch_bounds__(128) void k_qkv(const _Float16* __restrict__ xw,
                                             const _Float16* __restrict__ wt,
                                             const float* __restrict__ qkv_b,
                                             _Float16* __restrict__ q,
                                             _Float16* __restrict__ k,
                                             _Float16* __restrict__ v) {
  int wv = threadIdx.x >> 5, lane = threadIdx.x & 31;
  int id = blockIdx.x * 4 + wv;              // 0..5631
  int win = id / NT_TILES, mt = id % NT_TILES;
  const _Float16* A = xw + ((size_t)win * NPAD + mt * 16) * DIM;
  v16h a[4];
  #pragma unroll
  for (int kk = 0; kk < 4; ++kk) a[kk] = load_a_frag(A + kk * 32, DIM, lane);
  int col = lane & 15;
  int rowb = mt * 16 + (lane >> 4) * 8;
  bool full = (mt < NT_TILES - 1);           // all 16 rows valid (uniform per wave)
  for (int nt = 0; nt < 24; ++nt) {
    if (nt < 23) __builtin_prefetch(wt + (size_t)(nt + 1) * 16 * DIM, 0, 3);
    v8f acc = zero8();
    #pragma unroll
    for (int kk = 0; kk < 4; ++kk)
      acc = wmma16(a[kk], load_b_frag(wt + (size_t)nt * 16 * DIM + kk * 32, DIM, lane), acc);
    int f = nt * 16 + col;
    float bias = qkv_b[f];
    if (nt < 8) {                            // -> q (scaled)
      int hh = f >> 5, d = f & 31;
      _Float16* dst = q + (((size_t)win * NH + hh) * NPAD + rowb) * HD + d;
      if (full) {
        #pragma unroll
        for (int r = 0; r < 8; ++r) dst[(size_t)r * HD] = (_Float16)((acc[r] + bias) * QK_SCALE);
      } else {
        #pragma unroll
        for (int r = 0; r < 8; ++r)
          if (rowb + r < NTOK) dst[(size_t)r * HD] = (_Float16)((acc[r] + bias) * QK_SCALE);
      }
    } else if (nt < 16) {                    // -> k
      int f2 = f - DIM; int hh = f2 >> 5, d = f2 & 31;
      _Float16* dst = k + (((size_t)win * NH + hh) * NPAD + rowb) * HD + d;
      if (full) {
        #pragma unroll
        for (int r = 0; r < 8; ++r) dst[(size_t)r * HD] = (_Float16)(acc[r] + bias);
      } else {
        #pragma unroll
        for (int r = 0; r < 8; ++r)
          if (rowb + r < NTOK) dst[(size_t)r * HD] = (_Float16)(acc[r] + bias);
      }
    } else {                                 // -> v (transposed [h][d][tok])
      int f2 = f - 2 * DIM; int hh = f2 >> 5, d = f2 & 31;
      _Float16* dst = v + (((size_t)win * NH + hh) * HD + d) * NPAD + rowb;
      if (full) {
        #pragma unroll
        for (int r = 0; r < 8; ++r) dst[r] = (_Float16)(acc[r] + bias);
      } else {
        #pragma unroll
        for (int r = 0; r < 8; ++r)
          if (rowb + r < NTOK) dst[r] = (_Float16)(acc[r] + bias);
      }
    }
  }
}

// region id inside the shifted image (for the attention mask)
__device__ inline int region3(int gz, int gh, int gw) {
  int zi = gz < (ZP - WS) ? 0 : (gz < (ZP - SS) ? 1 : 2);
  int hi = gh < (HP - WS) ? 0 : (gh < (HP - SS) ? 1 : 2);
  int wi = gw < (WP - WS) ? 0 : (gw < (WP - SS) ? 1 : 2);
  return (zi * 3 + hi) * 3 + wi;
}

// ---------------- attention: scores + bias + mask + softmax + P*V -------------
__global__ __launch_bounds__(32) void k_attn(const _Float16* __restrict__ q,
                                             const _Float16* __restrict__ k,
                                             const _Float16* __restrict__ v,
                                             const float* __restrict__ rel_table,
                                             _Float16* __restrict__ ao) {
  __shared__ __align__(32) _Float16 lp[16 * NPAD];
  int blk = blockIdx.x;                      // 256*4*22
  int mt = blk % NT_TILES;
  int whid = blk / NT_TILES;
  int head = whid % NH, win = whid / NH;
  int lane = threadIdx.x & 31;
  int col = lane & 15, hl = lane >> 4;

  const _Float16* qb = q + (((size_t)win * NH + head) * NPAD + mt * 16) * HD;
  v16h aq = load_a_frag(qb, HD, lane);

  float s[NT_TILES][8];
  #pragma unroll
  for (int nt = 0; nt < NT_TILES; ++nt) {
    v16h bk = load_b_frag(k + (((size_t)win * NH + head) * NPAD + nt * 16) * HD, HD, lane);
    v8f acc = zero8();
    acc = wmma16(aq, bk, acc);
    #pragma unroll
    for (int r = 0; r < 8; ++r) s[nt][r] = acc[r];
  }

  // bias + mask
  int wrem = win % 128;
  int wz = wrem / 64, wh = (wrem / 8) % 8, ww = wrem % 8;
  int rowb = mt * 16 + hl * 8;
  #pragma unroll
  for (int nt = 0; nt < NT_TILES; ++nt) {
    int m = nt * 16 + col;                   // key token
    #pragma unroll
    for (int r = 0; r < 8; ++r) {
      int n = rowb + r;                      // query token
      float val = s[nt][r];
      if (n < NTOK && m < NTOK) {
        int nz = n / 49, nh = (n / 7) % 7, nw = n % 7;
        int mz = m / 49, mh = (m / 7) % 7, mw = m % 7;
        int rel = ((nz - mz + 6) * 13 + (nh - mh + 6)) * 13 + (nw - mw + 6);
        val += rel_table[rel * NH + head];
        int regn = region3(wz * WS + nz, wh * WS + nh, ww * WS + nw);
        int regm = region3(wz * WS + mz, wh * WS + mh, ww * WS + mw);
        if (regn != regm) val -= 100.f;
      } else {
        val = -1e30f;
      }
      s[nt][r] = val;
    }
  }

  // row softmax (rows live across the 16 lanes of each half-wave)
  #pragma unroll
  for (int r = 0; r < 8; ++r) {
    float mx = -1e30f;
    #pragma unroll
    for (int nt = 0; nt < NT_TILES; ++nt) mx = fmaxf(mx, s[nt][r]);
    #pragma unroll
    for (int off = 8; off >= 1; off >>= 1) mx = fmaxf(mx, __shfl_xor(mx, off, 16));
    float sum = 0.f;
    #pragma unroll
    for (int nt = 0; nt < NT_TILES; ++nt) {
      float e = __expf(s[nt][r] - mx);
      s[nt][r] = e;
      sum += e;
    }
    #pragma unroll
    for (int off = 8; off >= 1; off >>= 1) sum += __shfl_xor(sum, off, 16);
    float inv = 1.f / sum;
    int lr = r + hl * 8;
    #pragma unroll
    for (int nt = 0; nt < NT_TILES; ++nt)
      lp[lr * NPAD + nt * 16 + col] = (_Float16)(s[nt][r] * inv);
  }
  __syncthreads();

  // P [16,352] @ V^T-layout [32][352]  -> out [16,32]
  const _Float16* vt = v + ((size_t)win * NH + head) * HD * NPAD;
  #pragma unroll
  for (int dt = 0; dt < 2; ++dt) {
    v8f acc = zero8();
    #pragma unroll
    for (int kk = 0; kk < 11; ++kk) {
      v16h ap = load_a_frag(&lp[kk * 32], NPAD, lane);
      v16h bv = load_b_frag(vt + (size_t)dt * 16 * NPAD + kk * 32, NPAD, lane);
      acc = wmma16(ap, bv, acc);
    }
    #pragma unroll
    for (int r = 0; r < 8; ++r) {
      int tok = mt * 16 + hl * 8 + r;
      ao[((size_t)win * NPAD + tok) * DIM + head * HD + dt * 16 + col] = (_Float16)acc[r];
    }
  }
}

// ---------------- proj GEMM + window reverse + un-shift + residual -> x1 ------
__global__ __launch_bounds__(128) void k_proj(const _Float16* __restrict__ ao,
                                              const _Float16* __restrict__ pwt,
                                              const float* __restrict__ proj_b,
                                              const float* __restrict__ x,
                                              float* __restrict__ x1) {
  int wv = threadIdx.x >> 5, lane = threadIdx.x & 31;
  int id = blockIdx.x * 4 + wv;              // 0..5631
  int win = id / NT_TILES, mt = id % NT_TILES;
  const _Float16* A = ao + ((size_t)win * NPAD + mt * 16) * DIM;
  v16h a[4];
  #pragma unroll
  for (int kk = 0; kk < 4; ++kk) a[kk] = load_a_frag(A + kk * 32, DIM, lane);
  int bb = win / 128, wrem = win % 128;
  int wz = wrem / 64, wh = (wrem / 8) % 8, ww = wrem % 8;
  int rowb = mt * 16 + (lane >> 4) * 8;
  bool full = (mt < NT_TILES - 1);
  for (int nt = 0; nt < 8; ++nt) {
    if (nt < 7) __builtin_prefetch(pwt + (size_t)(nt + 1) * 16 * DIM, 0, 3);
    v8f acc = zero8();
    #pragma unroll
    for (int kk = 0; kk < 4; ++kk)
      acc = wmma16(a[kk], load_b_frag(pwt + (size_t)nt * 16 * DIM + kk * 32, DIM, lane), acc);
    int c = nt * 16 + (lane & 15);
    float pb = proj_b[c];
    #pragma unroll
    for (int r = 0; r < 8; ++r) {
      int tok = rowb + r;
      if (!full && tok >= NTOK) continue;
      int tz = tok / 49, th = (tok / 7) % 7, tw = tok % 7;
      int z = (wz * WS + tz + SS) % ZP;
      int h = (wh * WS + th + SS) % HP;
      int w = (ww * WS + tw + SS) % WP;
      size_t idx = ((((size_t)bb * ZP + z) * HP + h) * WP + w) * DIM + c;
      x1[idx] = x[idx] + acc[r] + pb;
    }
  }
}

// ---------------- LN2 -> xn2 f16 ---------------------------------------------
__global__ __launch_bounds__(256) void k_ln2(const float* __restrict__ x1,
                                             const float* __restrict__ g,
                                             const float* __restrict__ b,
                                             _Float16* __restrict__ xn2) {
  int wv = threadIdx.x >> 5, lane = threadIdx.x & 31;
  int token = blockIdx.x * 8 + wv;
  const float* xp = x1 + (size_t)token * DIM;
  _Float16* dst = xn2 + (size_t)token * DIM;
  float v0[4], s = 0.f, s2 = 0.f;
  #pragma unroll
  for (int i = 0; i < 4; ++i) { v0[i] = xp[lane + 32 * i]; s += v0[i]; s2 += v0[i] * v0[i]; }
  #pragma unroll
  for (int off = 16; off >= 1; off >>= 1) {
    s  += __shfl_xor(s,  off, 32);
    s2 += __shfl_xor(s2, off, 32);
  }
  float mean = s * (1.f / DIM);
  float rstd = rsqrtf(s2 * (1.f / DIM) - mean * mean + 1e-5f);
  #pragma unroll
  for (int i = 0; i < 4; ++i) {
    int c = lane + 32 * i;
    dst[c] = (_Float16)((v0[i] - mean) * rstd * g[c] + b[c]);
  }
}

// ---------------- MLP fc1 + exact GELU; one mtile per wave --------------------
__global__ __launch_bounds__(128) void k_mlp1(const _Float16* __restrict__ xn2,
                                              const _Float16* __restrict__ w1t,
                                              const float* __restrict__ b1,
                                              _Float16* __restrict__ hbuf) {
  int wv = threadIdx.x >> 5, lane = threadIdx.x & 31;
  int mt = blockIdx.x * 4 + wv;              // 0..5487
  const _Float16* A = xn2 + (size_t)mt * 16 * DIM;
  v16h a[4];
  #pragma unroll
  for (int kk = 0; kk < 4; ++kk) a[kk] = load_a_frag(A + kk * 32, DIM, lane);
  int rowb = mt * 16 + (lane >> 4) * 8;
  for (int nt = 0; nt < 32; ++nt) {
    if (nt < 31) __builtin_prefetch(w1t + (size_t)(nt + 1) * 16 * DIM, 0, 3);
    v8f acc = zero8();
    #pragma unroll
    for (int kk = 0; kk < 4; ++kk)
      acc = wmma16(a[kk], load_b_frag(w1t + (size_t)nt * 16 * DIM + kk * 32, DIM, lane), acc);
    int c = nt * 16 + (lane & 15);
    float bb = b1[c];
    _Float16* dst = hbuf + (size_t)rowb * NMLP + c;
    #pragma unroll
    for (int r = 0; r < 8; ++r) {
      float xv = acc[r] + bb;
      float ge = 0.5f * xv * (1.f + erff(xv * 0.70710678118654752f));
      dst[(size_t)r * NMLP] = (_Float16)ge;
    }
  }
}

// ---------------- MLP fc2 + residual -> out; one mtile per wave ---------------
__global__ __launch_bounds__(128) void k_mlp2(const _Float16* __restrict__ hbuf,
                                              const _Float16* __restrict__ w2t,
                                              const float* __restrict__ b2,
                                              const float* __restrict__ x1,
                                              float* __restrict__ out) {
  int wv = threadIdx.x >> 5, lane = threadIdx.x & 31;
  int mt = blockIdx.x * 4 + wv;              // 0..5487
  const _Float16* A = hbuf + (size_t)mt * 16 * NMLP;
  v16h a[16];
  #pragma unroll
  for (int kk = 0; kk < 16; ++kk) a[kk] = load_a_frag(A + kk * 32, NMLP, lane);
  int rowb = mt * 16 + (lane >> 4) * 8;
  for (int nt = 0; nt < 8; ++nt) {
    if (nt < 7) __builtin_prefetch(w2t + (size_t)(nt + 1) * 16 * NMLP, 0, 3);
    v8f acc = zero8();
    #pragma unroll
    for (int kk = 0; kk < 16; ++kk)
      acc = wmma16(a[kk], load_b_frag(w2t + (size_t)nt * 16 * NMLP + kk * 32, NMLP, lane), acc);
    int c = nt * 16 + (lane & 15);
    float bb = b2[c];
    #pragma unroll
    for (int r = 0; r < 8; ++r) {
      size_t idx = ((size_t)(rowb + r)) * DIM + c;
      out[idx] = x1[idx] + acc[r] + bb;
    }
  }
}

// ---------------- scratch layout (bytes, 256-aligned) -------------------------
#define OFF_WT_QKV  ((size_t)0)
#define OFF_WT_PROJ ((size_t)98304)
#define OFF_WT_FC1  ((size_t)131072)
#define OFF_WT_FC2  ((size_t)262144)
#define OFF_XW      ((size_t)393216)
#define SZ_WINBUF   ((size_t)NWIN * NPAD * DIM * 2)     // 23,068,672
#define OFF_Q       (OFF_XW + SZ_WINBUF)
#define OFF_K       (OFF_Q + SZ_WINBUF)
#define OFF_V       (OFF_K + SZ_WINBUF)
#define OFF_AO      (OFF_V + SZ_WINBUF)
#define OFF_X1      (OFF_AO + SZ_WINBUF)
#define OFF_XN2     OFF_XW                               // reuse after attn path
#define OFF_H       OFF_Q                                // reuse q/k/v/ao span

extern "C" void kernel_launch(void* const* d_in, const int* in_sizes, int n_in,
                              void* d_out, int out_size, void* d_ws, size_t ws_size,
                              hipStream_t stream) {
  (void)in_sizes; (void)n_in; (void)out_size; (void)ws_size;
  const float* x      = (const float*)d_in[0];
  const float* n1g    = (const float*)d_in[1];
  const float* n1b    = (const float*)d_in[2];
  const float* qkv_w  = (const float*)d_in[3];
  const float* qkv_b  = (const float*)d_in[4];
  const float* relt   = (const float*)d_in[5];
  const float* proj_w = (const float*)d_in[6];
  const float* proj_b = (const float*)d_in[7];
  const float* n2g    = (const float*)d_in[8];
  const float* n2b    = (const float*)d_in[9];
  const float* fc1_w  = (const float*)d_in[10];
  const float* fc1_b  = (const float*)d_in[11];
  const float* fc2_w  = (const float*)d_in[12];
  const float* fc2_b  = (const float*)d_in[13];
  float* out = (float*)d_out;

  char* ws = (char*)d_ws;
  _Float16* wt_qkv  = (_Float16*)(ws + OFF_WT_QKV);
  _Float16* wt_proj = (_Float16*)(ws + OFF_WT_PROJ);
  _Float16* wt_fc1  = (_Float16*)(ws + OFF_WT_FC1);
  _Float16* wt_fc2  = (_Float16*)(ws + OFF_WT_FC2);
  _Float16* xw      = (_Float16*)(ws + OFF_XW);
  _Float16* qb      = (_Float16*)(ws + OFF_Q);
  _Float16* kb      = (_Float16*)(ws + OFF_K);
  _Float16* vb      = (_Float16*)(ws + OFF_V);
  _Float16* ao      = (_Float16*)(ws + OFF_AO);
  float*    x1      = (float*)   (ws + OFF_X1);
  _Float16* xn2     = (_Float16*)(ws + OFF_XN2);
  _Float16* hbuf    = (_Float16*)(ws + OFF_H);

  k_transpose_f16<<<(128 * 384 + 255) / 256, 256, 0, stream>>>(qkv_w, wt_qkv, 128, 384);
  k_transpose_f16<<<(128 * 128 + 255) / 256, 256, 0, stream>>>(proj_w, wt_proj, 128, 128);
  k_transpose_f16<<<(128 * 512 + 255) / 256, 256, 0, stream>>>(fc1_w, wt_fc1, 128, 512);
  k_transpose_f16<<<(512 * 128 + 255) / 256, 256, 0, stream>>>(fc2_w, wt_fc2, 512, 128);

  k_ln1<<<NWIN * NPAD / 8, 256, 0, stream>>>(x, n1g, n1b, xw);
  k_qkv<<<NWIN * NT_TILES / 4, 128, 0, stream>>>(xw, wt_qkv, qkv_b, qb, kb, vb);
  k_attn<<<NWIN * NH * NT_TILES, 32, 0, stream>>>(qb, kb, vb, relt, ao);
  k_proj<<<NWIN * NT_TILES / 4, 128, 0, stream>>>(ao, wt_proj, proj_b, x, x1);
  k_ln2<<<TOKENS / 8, 256, 0, stream>>>(x1, n2g, n2b, xn2);
  k_mlp1<<<MT_MLP / 4, 128, 0, stream>>>(xn2, wt_fc1, fc1_b, hbuf);
  k_mlp2<<<MT_MLP / 4, 128, 0, stream>>>(hbuf, wt_fc2, fc2_b, x1, out);
}
